// GaussianUpsampling_58265526337766
// MI455X (gfx1250) — matrix operand order repaired and checked by reading, not compiled, not run on hardware
//
#include <hip/hip_runtime.h>
#include <hip/hip_bf16.h>
#include <stdint.h>

typedef __attribute__((ext_vector_type(16))) _Float16 v16h;
typedef __attribute__((ext_vector_type(8)))  _Float16 v8h;
typedef __attribute__((ext_vector_type(8)))  float    v8f;

#define GB 16
#define GL 512
#define GD 768
#define GT 2048

// ---------------- kernel 1: gaussian centers via wave32 scan ----------------
__global__ void centers_kernel(const float* __restrict__ duration,
                               float* __restrict__ center) {
  const int b = blockIdx.x;
  const int lane = threadIdx.x;            // 32 threads (one wave32)
  const float* d = duration + (size_t)b * GL;
  float* c = center + (size_t)b * GL;
  const int CH = GL / 32;                  // 16 elements per lane
  float dv[16];
  float s = 0.f;
  const int base = lane * CH;
  #pragma unroll
  for (int i = 0; i < 16; ++i) { dv[i] = d[base + i]; s += dv[i]; }
  // inclusive scan of per-lane totals across the wave
  float incl = s;
  #pragma unroll
  for (int off = 1; off < 32; off <<= 1) {
    float n = __shfl_up(incl, off, 32);
    if (lane >= off) incl += n;
  }
  float run = incl - s;                    // exclusive prefix
  #pragma unroll
  for (int i = 0; i < 16; ++i) {
    run += dv[i];
    c[base + i] = run - 0.5f * dv[i];
  }
}

// ------------- kernel 2: memory [B,L,D] f32 -> memT [B,D,L] f16 -------------
__global__ void mem_transpose_kernel(const float* __restrict__ mem,
                                     _Float16* __restrict__ memT) {
  __shared__ _Float16 tile[32][33];        // +1 pad to dodge bank conflicts
  const int b  = blockIdx.z;
  const int l0 = blockIdx.x * 32;
  const int d0 = blockIdx.y * 32;
  const int tx = threadIdx.x & 31;
  const int ty = threadIdx.x >> 5;         // 0..7
  const float* src = mem + (size_t)b * GL * GD;
  #pragma unroll
  for (int r = 0; r < 4; ++r) {
    int l = l0 + ty + r * 8;
    tile[ty + r * 8][tx] = (_Float16)src[(size_t)l * GD + d0 + tx];
  }
  __syncthreads();
  _Float16* dst = memT + (size_t)b * GD * GL;
  #pragma unroll
  for (int r = 0; r < 4; ++r) {
    int dd = d0 + ty + r * 8;
    dst[(size_t)dd * GL + l0 + tx] = tile[tx][ty + r * 8];
  }
}

// ------ kernel 3: gaussian pdf + normalize; attn f32 [B,L,T] + attnT f16 ----
__global__ void attn_kernel(const float* __restrict__ center,
                            const float* __restrict__ sigma,
                            const unsigned char* __restrict__ mask,
                            float* __restrict__ attn_out,    // [B,L,T] f32
                            _Float16* __restrict__ attnT) {  // [B,T,L] f16
  __shared__ float sC[GL];
  __shared__ float sIS[GL];                // inv-sigma, zeroed where masked
  const int b = blockIdx.y;
  const int t = blockIdx.x * blockDim.x + threadIdx.x;
  for (int l = threadIdx.x; l < GL; l += blockDim.x) {
    sC[l] = center[(size_t)b * GL + l];
    float is = 1.0f / sigma[(size_t)b * GL + l];
    sIS[l] = mask[(size_t)b * GL + l] ? 0.0f : is;
  }
  __syncthreads();
  const float KN = 0.3989422804014327f;    // 1/sqrt(2*pi)
  const float tf = (float)t;
  float sum = 0.f;
  for (int l = 0; l < GL; ++l) {
    float is = sIS[l];                     // is==0 (masked): exp(0)*0 == 0, correct
    float z = (tf - sC[l]) * is;
    sum += __expf(-0.5f * z * z) * is;
  }
  const float inv = 1.0f / (sum * KN + 1e-8f);
  float*    po = attn_out + (size_t)b * GL * GT + t;             // stride T over l
  _Float16* pt = attnT + ((size_t)b * GT + t) * (size_t)GL;      // contiguous in l
  for (int l = 0; l < GL; ++l) {
    float is = sIS[l];
    float z = (tf - sC[l]) * is;
    float w = __expf(-0.5f * z * z) * is * KN * inv;
    po[(size_t)l * GT] = w;
    pt[l] = (_Float16)w;
  }
}

// ------- kernel 4: batched GEMM out[b,t,d] = sum_l attnT[b,t,l]*mem[b,l,d] --
// A = attnT [T,L] f16 (row-major, K contiguous), B accessed via memT [D,L]
// (i.e. B^T, K contiguous). Each wave computes a 16x64 strip of C with
// v_wmma_f32_16x16x32_f16; 8 waves/block -> 128x64 macro tile.
__global__ void __launch_bounds__(256)
upsample_wmma_kernel(const _Float16* __restrict__ attnT,  // [B,T,L]
                     const _Float16* __restrict__ memT,   // [B,D,L]
                     float* __restrict__ out) {           // [B,T,D]
  const int b     = blockIdx.z;
  const int wave  = threadIdx.x >> 5;
  const int lane  = threadIdx.x & 31;
  const int mBase = blockIdx.x * 128 + wave * 16;
  const int nBase = blockIdx.y * 64;
  const int l16   = lane & 15;
  const int hiH   = lane >> 4;             // 0: lanes 0-15, 1: lanes 16-31

  // A fragment: row m = l16, 16 contiguous K starting at kb = 16*hiH
  const _Float16* Arow = attnT + ((size_t)b * GT + mBase + l16) * GL + hiH * 16;
  // B fragment: col n = l16, 16 contiguous K starting at kh = 16*hiH
  const _Float16* Bcol = memT + ((size_t)b * GD + nBase + l16) * GL + hiH * 16;

  v8f acc[4] = {};
  for (int k0 = 0; k0 < GL; k0 += 32) {
    v8h alo = *(const v8h*)(Arow + k0);        // K = kb .. kb+7
    v8h ahi = *(const v8h*)(Arow + k0 + 8);    // K = kb+8 .. kb+15
    v16h a = __builtin_shufflevector(alo, ahi,
               0,1,2,3,4,5,6,7,8,9,10,11,12,13,14,15);
    __builtin_prefetch(Arow + k0 + 32, 0, 1);  // global_prefetch_b8
    #pragma unroll
    for (int nt = 0; nt < 4; ++nt) {
      v16h bf = *(const v16h*)(Bcol + (size_t)nt * 16 * GL + k0);
      acc[nt] = __builtin_amdgcn_wmma_f32_16x16x32_f16(
          /*neg_a=*/false, a, /*neg_b=*/false, bf,
          /*c_mod=*/(short)0, acc[nt], /*reuse_a=*/false, /*reuse_b=*/false);
    }
  }
  // C/D layout: VGPR r -> row r (lanes 0-15) / row r+8 (lanes 16-31), col = lane&15
  const int col = nBase + l16;
  const int row0 = mBase + hiH * 8;
  #pragma unroll
  for (int nt = 0; nt < 4; ++nt) {
    float* o = out + ((size_t)b * GT + row0) * GD + col + nt * 16;
    #pragma unroll
    for (int r = 0; r < 8; ++r)
      o[(size_t)r * GD] = acc[nt][r];
  }
}

extern "C" void kernel_launch(void* const* d_in, const int* in_sizes, int n_in,
                              void* d_out, int out_size, void* d_ws, size_t ws_size,
                              hipStream_t stream) {
  (void)in_sizes; (void)n_in; (void)out_size; (void)ws_size;

  const float* memory   = (const float*)d_in[0];          // [B,L,D] f32
  const float* duration = (const float*)d_in[1];          // [B,L]   f32
  const float* sigma    = (const float*)d_in[2];          // [B,L]   f32
  // d_in[3] = output_lengths (all == T_MAX, unused)
  const unsigned char* mask = (const unsigned char*)d_in[4]; // [B,L] bool

  // workspace layout
  char* ws = (char*)d_ws;
  float*    center = (float*)ws;                                   // B*L*4
  _Float16* memT   = (_Float16*)(ws + 32768);                      // B*D*L*2
  _Float16* attnT  = (_Float16*)(ws + 32768 + (size_t)GB*GD*GL*2); // B*T*L*2

  // outputs, concatenated: upsampled [B,T,D] f32 then attn_weights [B,L,T] f32
  float* upsampled = (float*)d_out;
  float* attn_out  = upsampled + (size_t)GB * GT * GD;

  centers_kernel<<<GB, 32, 0, stream>>>(duration, center);
  mem_transpose_kernel<<<dim3(GL/32, GD/32, GB), 256, 0, stream>>>(memory, memT);
  attn_kernel<<<dim3(GT/256, GB), 256, 0, stream>>>(center, sigma, mask,
                                                    attn_out, attnT);
  upsample_wmma_kernel<<<dim3(GT/128, GD/64, GB), 256, 0, stream>>>(attnT, memT,
                                                                    upsampled);
}
